// PhysicsNet_56959856279613
// MI455X (gfx1250) — matrix operand order repaired and checked by reading, not compiled, not run on hardware
//
#include <hip/hip_runtime.h>
#include <hip/hip_bf16.h>
#include <math.h>

// ---------------------------------------------------------------------------
// PhysicsNet for MI455X (gfx1250, wave32, WMMA).
// All dense layers + attention run through WMMA GEMM kernels using
// v_wmma_f32_16x16x32_f16.  The hot GEMMs (full 64x128x32 tiles) use the
// CDNA5 async-to-LDS path (GLOBAL_LOAD_ASYNC_TO_LDS_B128, ASYNCcnt,
// ping-pong double buffering) issued via inline asm; the irregular GEMMs
// (K=14 / N=32) use a bounds-checked fallback.  Neighbor searches
// (FPS / kNN / 3-NN) are VALU kernels with LDS reductions.  Working set
// (~60MB) is L2-resident (192MB L2): compute/latency bound, not HBM bound.
// ---------------------------------------------------------------------------

#define cB  2
#define cN  65536
#define cG  512
#define cK  32
#define cC  11
#define cD  128
#define cNH 4
#define cE  3
#define cCIN 14     // C + 3
#define EPSF 1e-5f

typedef __attribute__((ext_vector_type(16))) _Float16 v16h;
typedef __attribute__((ext_vector_type(8)))  float    v8f;

// ---------------------------------------------------------------------------
// Shared epilogue: v = acc*alpha (+bias) (BN) (act) (+res)
// ---------------------------------------------------------------------------
__device__ __forceinline__ float gemm_epilogue(float v, int gn, float alpha,
                                               const float* bias,
                                               const float* bn_m, const float* bn_v,
                                               const float* bn_g, const float* bn_b,
                                               int act)
{
  v *= alpha;
  if (bias) v += bias[gn];
  if (bn_m) v = (v - bn_m[gn]) / sqrtf(bn_v[gn] + EPSF) * bn_g[gn] + bn_b[gn];
  if (act == 1)      v = fmaxf(v, 0.f);
  else if (act == 2) v = 0.5f * v * (1.f + erff(v * 0.70710678118654752f));
  return v;
}

// ---------------------------------------------------------------------------
// Generic bounds-checked WMMA GEMM (fallback): BM=BN=64, BK=32, 8 waves,
// 2 output tiles per wave, f16 tiles staged in LDS with scalar guards.
// ---------------------------------------------------------------------------
#define BM 64
#define BN 64
#define BKK 32

__global__ __launch_bounds__(256) void wmma_gemm_kernel(
    const float* __restrict__ A, int lda, long aO, long aI,
    const float* __restrict__ Bm, int ldb, long bO, long bI, int transB,
    float* __restrict__ Cm, int ldc, long cO, long cI,
    const float* __restrict__ Res, long rO, long rI,
    int innerCount, int M, int N, int Kd, float alpha,
    const float* __restrict__ bias,
    const float* __restrict__ bn_m, const float* __restrict__ bn_v,
    const float* __restrict__ bn_g, const float* __restrict__ bn_b,
    int act)
{
  __shared__ _Float16 sA[BM][BKK + 4];
  __shared__ _Float16 sB[BN][BKK + 4];

  const int tid  = threadIdx.x;
  const int z    = blockIdx.z;
  const int zo   = z / innerCount, zi = z % innerCount;
  const float* Ab = A  + (long)zo * aO + (long)zi * aI;
  const float* Bb = Bm + (long)zo * bO + (long)zi * bI;
  float*       Cb = Cm + (long)zo * cO + (long)zi * cI;
  const float* Rb = Res ? (Res + (long)zo * rO + (long)zi * rI) : nullptr;

  const int m0 = blockIdx.y * BM;
  const int n0 = blockIdx.x * BN;
  const int wave = tid >> 5, lane = tid & 31;
  const int hlf  = lane >> 4, l16 = lane & 15;
  const int tm   = wave & 3;          // M-tile 0..3
  const int tn0  = (wave >> 2) * 2;   // N-tile base 0 or 2

  v8f acc0 = {}; v8f acc1 = {};

  for (int k0 = 0; k0 < Kd; k0 += BKK) {
    for (int idx = tid; idx < BM * BKK; idx += 256) {
      int r = idx >> 5, c = idx & 31;
      int gm = m0 + r, gk = k0 + c;
      float v = (gm < M && gk < Kd) ? Ab[(long)gm * lda + gk] : 0.f;
      sA[r][c] = (_Float16)v;
    }
    for (int idx = tid; idx < BN * BKK; idx += 256) {
      int r = idx >> 5, c = idx & 31;
      int gn = n0 + r, gk = k0 + c;
      float v = 0.f;
      if (gn < N && gk < Kd)
        v = transB ? Bb[(long)gn * ldb + gk] : Bb[(long)gk * ldb + gn];
      sB[r][c] = (_Float16)v;
    }
    if (k0 + BKK < Kd && m0 + (tid >> 5) < M)
      __builtin_prefetch(&Ab[(long)(m0 + (tid >> 5)) * lda + (k0 + BKK)], 0, 1);
    __syncthreads();

    v16h afrag, bfrag0, bfrag1;
    const int am = tm * 16 + l16;
#pragma unroll
    for (int e = 0; e < 16; ++e) {   // A 16x32: k = 16*hlf + e (matches B)
      int kk = hlf * 16 + e;
      afrag[e] = sA[am][kk];
    }
    const int bn0i = (tn0 + 0) * 16 + l16;
    const int bn1i = (tn0 + 1) * 16 + l16;
#pragma unroll
    for (int e = 0; e < 16; ++e) {   // B 32x16: k = 16*hlf + e (matches A)
      int kk = hlf * 16 + e;
      bfrag0[e] = sB[bn0i][kk];
      bfrag1[e] = sB[bn1i][kk];
    }
    acc0 = __builtin_amdgcn_wmma_f32_16x16x32_f16(false, afrag, false, bfrag0,
                                                  (short)0, acc0, false, false);
    acc1 = __builtin_amdgcn_wmma_f32_16x16x32_f16(false, afrag, false, bfrag1,
                                                  (short)0, acc1, false, false);
    __syncthreads();
  }

#pragma unroll
  for (int r = 0; r < 8; ++r) {
    int gm = m0 + tm * 16 + r + 8 * hlf;
    if (gm >= M) continue;
#pragma unroll
    for (int t = 0; t < 2; ++t) {
      int gn = n0 + (tn0 + t) * 16 + l16;
      if (gn >= N) continue;
      float v = gemm_epilogue(t == 0 ? acc0[r] : acc1[r], gn, alpha,
                              bias, bn_m, bn_v, bn_g, bn_b, act);
      if (Rb) v += Rb[(long)gm * ldc + gn];
      Cb[(long)gm * ldc + gn] = v;
    }
  }
}

// ---------------------------------------------------------------------------
// Fast WMMA GEMM with CDNA5 async-to-LDS double buffering.
// Requires M%64==0, N%128==0, K%32==0 (all hot GEMMs qualify).
// BM=64, BN=128, BK=32; 8 waves, 4 output tiles per wave (A frag reused).
// Tiles staged as f32 by GLOBAL_LOAD_ASYNC_TO_LDS_B128 (ASYNCcnt-tracked,
// per-lane LDS destinations make the B transpose free); fragment builds
// convert to f16.  Exactly 6 async instructions per wave per tile
// (A: 512 chunks / 256 thr = 2;  B: 1024 / 256 = 4), so the pipelined wait
// is  s_wait_asynccnt 6  (prev tile drained, next tile in flight).
// ---------------------------------------------------------------------------
#define BNA 128
#define A_PITCH 36     // 64x32 tile, +4 pad (144B rows: 16B aligned)
#define BT_PITCH 36    // B^T layout [n][k]
#define BK_PITCH 132   // B layout [k][n] (528B rows: 16B aligned)

__device__ __forceinline__ void async_b128(unsigned lds_byte, unsigned voff_byte,
                                           const float* base)
{
  asm volatile("global_load_async_to_lds_b128 %0, %1, %2"
               :: "v"(lds_byte), "v"(voff_byte),
                  "s"((unsigned long long)(uintptr_t)base)
               : "memory");
}

__device__ __forceinline__ void stage_tile_async(
    float* sA, float* sB,
    const float* Ab, int lda, const float* Bb, int ldb, int transB,
    int m0, int n0, int k0, int tid)
{
  // A tile: 64x32 f32, lds row pitch A_PITCH
#pragma unroll
  for (int i = 0; i < 2; ++i) {
    int chunk = tid + i * 256;                 // 0..511 (4 floats each)
    int r = chunk >> 3, c4 = (chunk & 7) * 4;
    unsigned lds  = (unsigned)(uintptr_t)(const void*)&sA[r * A_PITCH + c4];
    unsigned voff = (unsigned)(((long)(m0 + r) * lda + (k0 + c4)) * 4);
    async_b128(lds, voff, Ab);
  }
  if (transB) {
    // B stored N-major in memory: stage rows directly as sB[n][k]
#pragma unroll
    for (int i = 0; i < 4; ++i) {
      int chunk = tid + i * 256;               // 0..1023
      int n = chunk >> 3, c4 = (chunk & 7) * 4;
      unsigned lds  = (unsigned)(uintptr_t)(const void*)&sB[n * BT_PITCH + c4];
      unsigned voff = (unsigned)(((long)(n0 + n) * ldb + (k0 + c4)) * 4);
      async_b128(lds, voff, Bb);
    }
  } else {
    // B stored K-major: stage as sB[k][n] (contiguous along n)
#pragma unroll
    for (int i = 0; i < 4; ++i) {
      int chunk = tid + i * 256;
      int k = chunk >> 5, n4 = (chunk & 31) * 4;
      unsigned lds  = (unsigned)(uintptr_t)(const void*)&sB[k * BK_PITCH + n4];
      unsigned voff = (unsigned)(((long)(k0 + k) * ldb + (n0 + n4)) * 4);
      async_b128(lds, voff, Bb);
    }
  }
}

__global__ __launch_bounds__(256) void wmma_gemm_async_kernel(
    const float* __restrict__ A, int lda, long aO, long aI,
    const float* __restrict__ Bm, int ldb, long bO, long bI, int transB,
    float* __restrict__ Cm, int ldc, long cO, long cI,
    const float* __restrict__ Res, long rO, long rI,
    int innerCount, int M, int N, int Kd, float alpha,
    const float* __restrict__ bias,
    const float* __restrict__ bn_m, const float* __restrict__ bn_v,
    const float* __restrict__ bn_g, const float* __restrict__ bn_b,
    int act)
{
  __shared__ float sAf[2][64 * A_PITCH];      // 18 KB
  __shared__ float sBf[2][128 * BT_PITCH];    // 36 KB (covers both B layouts)

  const int tid = threadIdx.x;
  const int z   = blockIdx.z;
  const int zo  = z / innerCount, zi = z % innerCount;
  const float* Ab = A  + (long)zo * aO + (long)zi * aI;
  const float* Bb = Bm + (long)zo * bO + (long)zi * bI;
  float*       Cb = Cm + (long)zo * cO + (long)zi * cI;
  const float* Rb = Res ? (Res + (long)zo * rO + (long)zi * rI) : nullptr;

  const int m0 = blockIdx.y * 64;
  const int n0 = blockIdx.x * BNA;
  const int wave = tid >> 5, lane = tid & 31;
  const int hlf  = lane >> 4, l16 = lane & 15;
  const int tm   = wave & 3;                  // M-tile 0..3
  const int tnb  = (wave >> 2) * 4;           // N-tile base 0 or 4

  v8f acc[4] = {};

  stage_tile_async(sAf[0], sBf[0], Ab, lda, Bb, ldb, transB, m0, n0, 0, tid);

  int buf = 0;
  for (int k0 = 0; k0 < Kd; k0 += BKK, buf ^= 1) {
    if (k0 + BKK < Kd) {
      stage_tile_async(sAf[buf ^ 1], sBf[buf ^ 1], Ab, lda, Bb, ldb, transB,
                       m0, n0, k0 + BKK, tid);
      asm volatile("s_wait_asynccnt 6" ::: "memory");  // prev tile complete
    } else {
      asm volatile("s_wait_asynccnt 0" ::: "memory");
    }
    __syncthreads();

    const float* sA = sAf[buf];
    const float* sB = sBf[buf];
    v16h afrag;
    const int am = tm * 16 + l16;
#pragma unroll
    for (int e = 0; e < 16; ++e) {   // A 16x32: k = 16*hlf + e (matches B)
      int kk = hlf * 16 + e;
      afrag[e] = (_Float16)sA[am * A_PITCH + kk];
    }
#pragma unroll
    for (int t = 0; t < 4; ++t) {
      const int bn = (tnb + t) * 16 + l16;
      v16h bfrag;
      if (transB) {
#pragma unroll
        for (int e = 0; e < 16; ++e)
          bfrag[e] = (_Float16)sB[bn * BT_PITCH + hlf * 16 + e];
      } else {
#pragma unroll
        for (int e = 0; e < 16; ++e)
          bfrag[e] = (_Float16)sB[(hlf * 16 + e) * BK_PITCH + bn];
      }
      acc[t] = __builtin_amdgcn_wmma_f32_16x16x32_f16(false, afrag, false, bfrag,
                                                      (short)0, acc[t], false, false);
    }
    __syncthreads();   // readers done before this buffer is refilled
  }

#pragma unroll
  for (int r = 0; r < 8; ++r) {
    int gm = m0 + tm * 16 + r + 8 * hlf;
#pragma unroll
    for (int t = 0; t < 4; ++t) {
      int gn = n0 + (tnb + t) * 16 + l16;
      float v = gemm_epilogue(acc[t][r], gn, alpha, bias, bn_m, bn_v, bn_g, bn_b, act);
      if (Rb) v += Rb[(long)gm * ldc + gn];
      Cb[(long)gm * ldc + gn] = v;
    }
  }
}

// ---------------------------------------------------------------------------
// Farthest point sampling: one 1024-thread block per batch; min-dist array in
// registers (64/thread), per-round LDS argmax tree.  Emits center positions.
// ---------------------------------------------------------------------------
__global__ __launch_bounds__(1024) void fps_kernel(const float* __restrict__ pos,
                                                   float* __restrict__ centers)
{
  const int b = blockIdx.x;
  const int tid = threadIdx.x;
  const float* pb = pos + (long)b * cN * 3;
  float md[64];
#pragma unroll
  for (int j = 0; j < 64; ++j) md[j] = 1e10f;

  __shared__ float sv[1024];
  __shared__ int   si[1024];
  __shared__ int   s_last;
  int last = 0;
  for (int g = 0; g < cG; ++g) {
    float px = pb[last * 3 + 0], py = pb[last * 3 + 1], pz = pb[last * 3 + 2];
    if (tid == 0) {
      float* c = centers + ((long)b * cG + g) * 3;
      c[0] = px; c[1] = py; c[2] = pz;
    }
    float bestv = -1.f; int besti = 0;
#pragma unroll
    for (int j = 0; j < 64; ++j) {
      int i = tid + j * 1024;
      float dx = pb[i * 3 + 0] - px, dy = pb[i * 3 + 1] - py, dz = pb[i * 3 + 2] - pz;
      float d = dx * dx + dy * dy + dz * dz;
      float m = fminf(md[j], d);
      md[j] = m;
      if (m > bestv) { bestv = m; besti = i; }
    }
    sv[tid] = bestv; si[tid] = besti;
    __syncthreads();
    for (int s = 512; s > 0; s >>= 1) {
      if (tid < s && sv[tid + s] > sv[tid]) { sv[tid] = sv[tid + s]; si[tid] = si[tid + s]; }
      __syncthreads();
    }
    if (tid == 0) s_last = si[0];
    __syncthreads();
    last = s_last;
  }
}

// ---------------------------------------------------------------------------
// kNN grouping: one 128-thread block per (b,g).  Per-thread replace-max top-32
// (register arrays, unrolled dynamic index), LDS merge by 32 argmin rounds,
// then the 32 selected points are gathered into gi = [dpos | feat] rows.
// Neighbor ORDER is irrelevant (max-pool downstream), so no sort.
// ---------------------------------------------------------------------------
__global__ __launch_bounds__(128) void knn_group_kernel(
    const float* __restrict__ pos, const float* __restrict__ feat,
    const float* __restrict__ centers, float* __restrict__ gi)
{
  const int bg = blockIdx.x;
  const int b  = bg / cG;
  const int tid = threadIdx.x;
  const float* pb = pos  + (long)b * cN * 3;
  const float* fb = feat + (long)b * cN * cC;
  const float cx = centers[(long)bg * 3 + 0];
  const float cy = centers[(long)bg * 3 + 1];
  const float cz = centers[(long)bg * 3 + 2];

  float bd[32]; int bi[32];
#pragma unroll
  for (int j = 0; j < 32; ++j) { bd[j] = 3e38f; bi[j] = 0; }
  float worst = 3e38f; int wslot = 0;

  for (int i = tid; i < cN; i += 128) {
    float dx = pb[i * 3 + 0] - cx, dy = pb[i * 3 + 1] - cy, dz = pb[i * 3 + 2] - cz;
    float d = dx * dx + dy * dy + dz * dz;
    if (d < worst) {
#pragma unroll
      for (int j = 0; j < 32; ++j) if (j == wslot) { bd[j] = d; bi[j] = i; }
      float w = -1.f; int ws = 0;
#pragma unroll
      for (int j = 0; j < 32; ++j) if (bd[j] > w) { w = bd[j]; ws = j; }
      worst = w; wslot = ws;
    }
  }

  __shared__ float sd[128 * 32];
  __shared__ int   sx[128 * 32];
  __shared__ float rv[128];
  __shared__ int   ri[128];
  __shared__ int   sel[32];
#pragma unroll
  for (int j = 0; j < 32; ++j) { sd[tid * 32 + j] = bd[j]; sx[tid * 32 + j] = bi[j]; }
  __syncthreads();

  for (int k = 0; k < 32; ++k) {
    float mv = 3e38f; int me = 0;
    for (int e = tid; e < 128 * 32; e += 128)
      if (sd[e] < mv) { mv = sd[e]; me = e; }
    rv[tid] = mv; ri[tid] = me;
    __syncthreads();
    for (int s = 64; s > 0; s >>= 1) {
      if (tid < s && rv[tid + s] < rv[tid]) { rv[tid] = rv[tid + s]; ri[tid] = ri[tid + s]; }
      __syncthreads();
    }
    if (tid == 0) { int e = ri[0]; sel[k] = sx[e]; sd[e] = 3e38f; }
    __syncthreads();
  }

  if (tid < 32) {
    int pi = sel[tid];
    float* row = gi + ((long)bg * cK + tid) * cCIN;
    row[0] = pb[pi * 3 + 0] - cx;
    row[1] = pb[pi * 3 + 1] - cy;
    row[2] = pb[pi * 3 + 2] - cz;
#pragma unroll
    for (int c = 0; c < cC; ++c) row[3 + c] = fb[(long)pi * cC + c];
  }
}

// ---- group encoder: max over K + concat [h1 | hg] ----
__global__ __launch_bounds__(64) void concat_pool_kernel(const float* __restrict__ h1,
                                                         float* __restrict__ h1cat)
{
  const int bg = blockIdx.x, c = threadIdx.x; // 64 channels
  const float* base = h1 + (long)bg * cK * 64;
  float m = -3e38f;
  for (int k = 0; k < cK; ++k) m = fmaxf(m, base[k * 64 + c]);
  float* ob = h1cat + (long)bg * cK * cD;
  for (int k = 0; k < cK; ++k) {
    ob[k * cD + c]      = base[k * 64 + c];
    ob[k * cD + 64 + c] = m;
  }
}

__global__ __launch_bounds__(128) void maxpool_gf_kernel(const float* __restrict__ h2,
                                                         float* __restrict__ gf)
{
  const int bg = blockIdx.x, c = threadIdx.x; // 128 channels
  const float* base = h2 + (long)bg * cK * cD;
  float m = -3e38f;
  for (int k = 0; k < cK; ++k) m = fmaxf(m, base[k * cD + c]);
  gf[(long)bg * cD + c] = m;
}

// ---- positional embedding layer 1 (K=3, scalar) ----
__global__ __launch_bounds__(128) void pe1_kernel(const float* __restrict__ centers,
                                                  const float* __restrict__ w1,
                                                  const float* __restrict__ b1,
                                                  float* __restrict__ pe1)
{
  const int bg = blockIdx.x, c = threadIdx.x;
  float s = b1[c];
#pragma unroll
  for (int j = 0; j < 3; ++j) s += centers[(long)bg * 3 + j] * w1[j * cD + c];
  pe1[(long)bg * cD + c] = fmaxf(s, 0.f);
}

// ---- LayerNorm over D=128 (one row per 128-thread block) ----
__global__ __launch_bounds__(128) void ln_kernel(const float* __restrict__ x,
                                                 const float* __restrict__ g,
                                                 const float* __restrict__ b,
                                                 float* __restrict__ y)
{
  const int row = blockIdx.x, t = threadIdx.x;
  float v = x[(long)row * cD + t];
  __shared__ float red[128];
  red[t] = v; __syncthreads();
  for (int s = 64; s > 0; s >>= 1) { if (t < s) red[t] += red[t + s]; __syncthreads(); }
  float mean = red[0] / 128.f;
  __syncthreads();
  float d = v - mean;
  red[t] = d * d; __syncthreads();
  for (int s = 64; s > 0; s >>= 1) { if (t < s) red[t] += red[t + s]; __syncthreads(); }
  float var = red[0] / 128.f;
  y[(long)row * cD + t] = d / sqrtf(var + EPSF) * g[t] + b[t];
}

// ---- row softmax over 512 (attention scores, in-place) ----
__global__ __launch_bounds__(256) void softmax_kernel(float* __restrict__ s)
{
  const long row = blockIdx.x;
  float* p = s + row * (long)cG;
  const int t = threadIdx.x;
  float a = p[t], b = p[t + 256];
  __shared__ float red[256];
  red[t] = fmaxf(a, b); __syncthreads();
  for (int st = 128; st > 0; st >>= 1) { if (t < st) red[t] = fmaxf(red[t], red[t + st]); __syncthreads(); }
  float M = red[0];
  __syncthreads();
  float ea = expf(a - M), eb = expf(b - M);
  red[t] = ea + eb; __syncthreads();
  for (int st = 128; st > 0; st >>= 1) { if (t < st) red[t] += red[t + st]; __syncthreads(); }
  float S = red[0];
  p[t] = ea / S; p[t + 256] = eb / S;
}

// ---- decoder output layer (N=3, LDS reduction) ----
__global__ __launch_bounds__(128) void dec_out_kernel(const float* __restrict__ h,
                                                      const float* __restrict__ w2,
                                                      const float* __restrict__ b2,
                                                      float* __restrict__ logits)
{
  const int row = blockIdx.x, t = threadIdx.x;
  float hv = h[(long)row * cD + t];
  __shared__ float red[128];
  for (int e = 0; e < cE; ++e) {
    red[t] = hv * w2[t * cE + e];
    __syncthreads();
    for (int s = 64; s > 0; s >>= 1) { if (t < s) red[t] += red[t + s]; __syncthreads(); }
    if (t == 0) logits[(long)row * cE + e] = red[0] + b2[e];
    __syncthreads();
  }
}

// ---- per-particle 3-NN over centers, logit mean, hard gumbel-softmax ----
__global__ __launch_bounds__(256) void assign_kernel(const float* __restrict__ pos,
                                                     const float* __restrict__ centers,
                                                     const float* __restrict__ logits,
                                                     const float* __restrict__ gum,
                                                     float* __restrict__ out)
{
  const int b = blockIdx.y;
  const int n = blockIdx.x * 256 + threadIdx.x;
  __shared__ float sc[cG * 3];
  __shared__ float sl[cG * cE];
  for (int i = threadIdx.x; i < cG * 3; i += 256) sc[i] = centers[(long)b * cG * 3 + i];
  for (int i = threadIdx.x; i < cG * cE; i += 256) sl[i] = logits[(long)b * cG * cE + i];
  __syncthreads();

  const long pn = (long)b * cN + n;
  float px = pos[pn * 3 + 0], py = pos[pn * 3 + 1], pz = pos[pn * 3 + 2];
  float d0 = 3e38f, d1 = 3e38f, d2 = 3e38f;
  int i0 = 0, i1 = 0, i2 = 0;
  for (int j = 0; j < cG; ++j) {
    float dx = sc[j * 3 + 0] - px, dy = sc[j * 3 + 1] - py, dz = sc[j * 3 + 2] - pz;
    float d = dx * dx + dy * dy + dz * dz;
    if (d < d2) {
      if (d < d1) {
        if (d < d0) { d2 = d1; i2 = i1; d1 = d0; i1 = i0; d0 = d; i0 = j; }
        else        { d2 = d1; i2 = i1; d1 = d;  i1 = j; }
      } else        { d2 = d;  i2 = j; }
    }
  }
  float z[cE];
#pragma unroll
  for (int e = 0; e < cE; ++e) {
    float pl = (sl[i0 * cE + e] + sl[i1 * cE + e] + sl[i2 * cE + e]) * (1.f / 3.f);
    float u = gum[pn * cE + e];
    u = fminf(fmaxf(u, 1e-9f), 1.f - 1e-9f);
    z[e] = pl - logf(-logf(u));
  }
  float m = fmaxf(z[0], fmaxf(z[1], z[2]));
  float e0 = expf(z[0] - m), e1 = expf(z[1] - m), e2 = expf(z[2] - m);
  float s = e0 + e1 + e2;
  float s0 = e0 / s, s1 = e1 / s, s2 = e2 / s;
  int am = 0; float bv = s0;
  if (s1 > bv) { bv = s1; am = 1; }
  if (s2 > bv) { bv = s2; am = 2; }
  float* o = out + pn * cE;
  o[0] = ((am == 0 ? 1.f : 0.f) - s0) + s0;
  o[1] = ((am == 1 ? 1.f : 0.f) - s1) + s1;
  o[2] = ((am == 2 ? 1.f : 0.f) - s2) + s2;
}

// ---------------------------------------------------------------------------
// Host side
// ---------------------------------------------------------------------------
static void gemm(hipStream_t st,
                 const float* A, int lda, long aO, long aI,
                 const float* Bm, int ldb, long bO, long bI, int transB,
                 float* C, int ldc, long cO, long cI,
                 const float* Res, long rO, long rI,
                 int inner, int batch, int M, int N, int Kd, float alpha,
                 const float* bias,
                 const float* bnm, const float* bnv, const float* bng, const float* bnb,
                 int act)
{
  const bool fast = (M % 64 == 0) && (N % BNA == 0) && (Kd % BKK == 0);
  if (fast) {
    dim3 grid(N / BNA, M / 64, batch);
    wmma_gemm_async_kernel<<<grid, dim3(256), 0, st>>>(
        A, lda, aO, aI, Bm, ldb, bO, bI, transB, C, ldc, cO, cI, Res, rO, rI,
        inner, M, N, Kd, alpha, bias, bnm, bnv, bng, bnb, act);
  } else {
    dim3 grid((N + BN - 1) / BN, (M + BM - 1) / BM, batch);
    wmma_gemm_kernel<<<grid, dim3(256), 0, st>>>(
        A, lda, aO, aI, Bm, ldb, bO, bI, transB, C, ldc, cO, cI, Res, rO, rI,
        inner, M, N, Kd, alpha, bias, bnm, bnv, bng, bnb, act);
  }
}

// Flattened input indices: [pos, features, <params pytree, sorted keys>, gumbel_u]
enum {
  IN_POS = 0, IN_FEAT = 1,
  P_DEC_B1 = 2, P_DEC_B2, P_DEC_LN_B, P_DEC_LN_G, P_DEC_W1, P_DEC_W2,
  P_GE_B1 = 8, P_GE_B2,
  P_GE_BN1_B = 10, P_GE_BN1_G, P_GE_BN1_M, P_GE_BN1_V,
  P_GE_BN2_B = 14, P_GE_BN2_G, P_GE_BN2_M, P_GE_BN2_V,
  P_GE_W1 = 18, P_GE_W2 = 19,
  P_L0 = 20,              // per-layer: b1,b2,bo,bqkv,ln1_b,ln1_g,ln2_b,ln2_g,w1,w2,wo,wqkv
  P_LSTRIDE = 12,
  P_PE_B1 = 44, P_PE_B2 = 45, P_PE_W1 = 46, P_PE_W2 = 47,
  IN_GUMBEL = 48
};

extern "C" void kernel_launch(void* const* d_in, const int* in_sizes, int n_in,
                              void* d_out, int out_size, void* d_ws, size_t ws_size,
                              hipStream_t stream)
{
  (void)in_sizes; (void)n_in; (void)out_size; (void)ws_size;
  auto P = [&](int i) { return (const float*)d_in[i]; };
  const float* pos  = P(IN_POS);
  const float* feat = P(IN_FEAT);
  const float* gum  = P(IN_GUMBEL);
  float* out = (float*)d_out;

  float* ws = (float*)d_ws;
  long off = 0;
  float* centers = ws + off; off += (long)cB * cG * 3;
  float* gi      = ws + off; off += (long)cB * cG * cK * cCIN;
  float* h1      = ws + off; off += (long)cB * cG * cK * 64;
  float* h1cat   = ws + off; off += (long)cB * cG * cK * cD;
  float* h2      = ws + off; off += (long)cB * cG * cK * cD;
  float* gf      = ws + off; off += (long)cB * cG * cD;
  float* pe1     = ws + off; off += (long)cB * cG * cD;
  float* x       = ws + off; off += (long)cB * cG * cD;
  float* y       = ws + off; off += (long)cB * cG * cD;
  float* qkv     = ws + off; off += (long)cB * cG * 3 * cD;
  float* smat    = ws + off; off += (long)cB * cNH * cG * cG;
  float* o       = ws + off; off += (long)cB * cG * cD;
  float* ffh     = ws + off; off += (long)cB * cG * 2 * cD;
  float* logits  = ws + off; off += (long)cB * cG * cE;

  const int BG = cB * cG;            // 1024 tokens
  const int MG = cB * cG * cK;       // 32768 group points

  // 1) FPS centers
  fps_kernel<<<cB, 1024, 0, stream>>>(pos, centers);

  // 2) kNN grouping + gather gi = [dpos | feat]
  knn_group_kernel<<<BG, 128, 0, stream>>>(pos, feat, centers, gi);

  // 3) Group encoder layer 1: relu(bn(gi @ ge_w1 + b1))   (K=14 -> fallback)
  gemm(stream, gi, cCIN, 0, 0, P(P_GE_W1), 64, 0, 0, 0,
       h1, 64, 0, 0, nullptr, 0, 0, 1, 1, MG, 64, cCIN, 1.f,
       P(P_GE_B1), P(P_GE_BN1_M), P(P_GE_BN1_V), P(P_GE_BN1_G), P(P_GE_BN1_B), 1);

  // 4) max over K + concat
  concat_pool_kernel<<<BG, 64, 0, stream>>>(h1, h1cat);

  // 5) Group encoder layer 2: relu(bn(h1cat @ ge_w2 + b2))   (async path)
  gemm(stream, h1cat, cD, 0, 0, P(P_GE_W2), cD, 0, 0, 0,
       h2, cD, 0, 0, nullptr, 0, 0, 1, 1, MG, cD, cD, 1.f,
       P(P_GE_B2), P(P_GE_BN2_M), P(P_GE_BN2_V), P(P_GE_BN2_G), P(P_GE_BN2_B), 1);

  // 6) gf = max over K
  maxpool_gf_kernel<<<BG, 128, 0, stream>>>(h2, gf);

  // 7) positional embedding; x = gf + pe
  pe1_kernel<<<BG, 128, 0, stream>>>(centers, P(P_PE_W1), P(P_PE_B1), pe1);
  gemm(stream, pe1, cD, 0, 0, P(P_PE_W2), cD, 0, 0, 0,
       x, cD, 0, 0, gf, 0, 0, 1, 1, BG, cD, cD, 1.f,
       P(P_PE_B2), nullptr, nullptr, nullptr, nullptr, 0);

  // 8) transformer layers
  const float inv_sqrt_dh = 1.0f / sqrtf(32.0f);
  for (int L = 0; L < 2; ++L) {
    const int pb = P_L0 + L * P_LSTRIDE;
    const float* b1   = P(pb + 0);
    const float* b2   = P(pb + 1);
    const float* bo   = P(pb + 2);
    const float* bqkv = P(pb + 3);
    const float* l1b  = P(pb + 4);
    const float* l1g  = P(pb + 5);
    const float* l2b  = P(pb + 6);
    const float* l2g  = P(pb + 7);
    const float* w1   = P(pb + 8);
    const float* w2   = P(pb + 9);
    const float* wo   = P(pb + 10);
    const float* wqkv = P(pb + 11);

    ln_kernel<<<BG, 128, 0, stream>>>(x, l1g, l1b, y);
    gemm(stream, y, cD, 0, 0, wqkv, 3 * cD, 0, 0, 0,
         qkv, 3 * cD, 0, 0, nullptr, 0, 0, 1, 1, BG, 3 * cD, cD, 1.f,
         bqkv, nullptr, nullptr, nullptr, nullptr, 0);

    // S = (Q @ K^T) / sqrt(dh)   per (b, h)   (async path, transB)
    gemm(stream, qkv, 3 * cD, (long)cG * 3 * cD, 32,
         qkv + cD, 3 * cD, (long)cG * 3 * cD, 32, 1,
         smat, cG, (long)cNH * cG * cG, (long)cG * cG,
         nullptr, 0, 0, cNH, cB * cNH, cG, cG, 32, inv_sqrt_dh,
         nullptr, nullptr, nullptr, nullptr, nullptr, 0);
    softmax_kernel<<<cB * cNH * cG, 256, 0, stream>>>(smat);

    // O = A @ V   per (b, h)   (N=32 -> fallback)
    gemm(stream, smat, cG, (long)cNH * cG * cG, (long)cG * cG,
         qkv + 2 * cD, 3 * cD, (long)cG * 3 * cD, 32, 0,
         o, cD, (long)cG * cD, 32,
         nullptr, 0, 0, cNH, cB * cNH, cG, 32, cG, 1.f,
         nullptr, nullptr, nullptr, nullptr, nullptr, 0);

    // x = x + O @ Wo + bo   (in place)
    gemm(stream, o, cD, 0, 0, wo, cD, 0, 0, 0,
         x, cD, 0, 0, x, 0, 0, 1, 1, BG, cD, cD, 1.f,
         bo, nullptr, nullptr, nullptr, nullptr, 0);

    // FFN
    ln_kernel<<<BG, 128, 0, stream>>>(x, l2g, l2b, y);
    gemm(stream, y, cD, 0, 0, w1, 2 * cD, 0, 0, 0,
         ffh, 2 * cD, 0, 0, nullptr, 0, 0, 1, 1, BG, 2 * cD, cD, 1.f,
         b1, nullptr, nullptr, nullptr, nullptr, 1);
    gemm(stream, ffh, 2 * cD, 0, 0, w2, cD, 0, 0, 0,
         x, cD, 0, 0, x, 0, 0, 1, 1, BG, cD, 2 * cD, 1.f,
         b2, nullptr, nullptr, nullptr, nullptr, 0);
  }

  // 9) expert decoder
  ln_kernel<<<BG, 128, 0, stream>>>(x, P(P_DEC_LN_G), P(P_DEC_LN_B), y);
  gemm(stream, y, cD, 0, 0, P(P_DEC_W1), cD, 0, 0, 0,
       ffh, cD, 0, 0, nullptr, 0, 0, 1, 1, BG, cD, cD, 1.f,
       P(P_DEC_B1), nullptr, nullptr, nullptr, nullptr, 2 /*gelu(erf)*/);
  dec_out_kernel<<<BG, 128, 0, stream>>>(ffh, P(P_DEC_W2), P(P_DEC_B2), logits);

  // 10) back to particles: 3-NN average + hard gumbel-softmax
  assign_kernel<<<dim3(cN / 256, cB), 256, 0, stream>>>(pos, centers, logits, gum, out);
}
